// EdgeBlock_5952824672852
// MI455X (gfx1250) — matrix operand order repaired and checked by reading, not compiled, not run on hardware
//
#include <hip/hip_runtime.h>
#include <hip/hip_bf16.h>

typedef __bf16 v16bf __attribute__((ext_vector_type(16)));
typedef float  v8f   __attribute__((ext_vector_type(8)));

#define DNODE 128
#define DHID  128
#define DOUT  128
#define TILEM 128            // edges per block
#define NWAVE 8
#define KC1   12             // 384 / 32 K-chunks, GEMM1
#define KC2   4              // 128 / 32 K-chunks, GEMM2
#define NT    8              // 128 / 16 N-tiles
#define W2OFF (KC1*NT*512)   // 49152 bf16 elements
#define WTOT  ((KC1+KC2)*NT*512) // 65536 bf16 elements = 128 KB

// round-to-nearest-even f32 -> bf16 bit pattern
__device__ __forceinline__ unsigned int f2bf(float f) {
  unsigned int u = __float_as_uint(f);
  u += 0x7FFFu + ((u >> 16) & 1u);
  return u >> 16;
}
__device__ __forceinline__ unsigned int pack2(float lo, float hi) {
  return f2bf(lo) | (f2bf(hi) << 16);
}

// ---------------------------------------------------------------------------
// Prep: swizzle W1 (384x128) and W2 (128x128) f32 -> bf16 into WMMA B-fragment
// layout.  B is 32x16 (KxN): lane%16 = N column; element e<8 -> K = 8*(lane/16)+e,
// e>=8 -> K = 16 + 8*(lane/16) + (e-8).
// Fragment f = kk*8 + nt; element address = (f*32 + lane)*16 + e.
// ---------------------------------------------------------------------------
__global__ void prep_weights(const float* __restrict__ W1,
                             const float* __restrict__ W2,
                             unsigned short* __restrict__ ws) {
  int idx = blockIdx.x * blockDim.x + threadIdx.x;
  const int n1 = W2OFF;
  if (idx >= WTOT) return;
  const float* W = (idx < n1) ? W1 : W2;
  int t = (idx < n1) ? idx : (idx - n1);
  int f = t >> 9, r = t & 511, lane = r >> 4, e = r & 15;
  int kk = f >> 3, nt = f & 7;
  int n = nt * 16 + (lane & 15);
  int k = kk * 32 + (e >> 3) * 16 + (lane >> 4) * 8 + (e & 7);
  ws[idx] = (unsigned short)f2bf(W[k * 128 + n]);
}

// ---------------------------------------------------------------------------
// Main: per block of 128 edges:
//   copy swizzled weights global->LDS (linear b128 copy),
//   gather [128 x 384] bf16 into LDS (A-fragment swizzle, b128 granularity),
//   GEMM1 (+b1, ReLU), in-LDS transpose, GEMM2 (+b2), store f32.
// ---------------------------------------------------------------------------
__global__ __launch_bounds__(256) void edge_mlp(
    const float* __restrict__ cell_attr, const float* __restrict__ edge_attr,
    const float* __restrict__ b1, const float* __restrict__ b2,
    const int* __restrict__ senders, const int* __restrict__ receivers,
    const unsigned short* __restrict__ wsw, float* __restrict__ out,
    int n_edges)
{
  __shared__ unsigned short ldsA[NWAVE * KC1 * 32 * 16]; // 96 KB, A-fragments
  __shared__ unsigned short ldsW[WTOT];                  // 128 KB, B-fragments

  const int tid  = threadIdx.x;
  const int lane = tid & 31;
  const int wave = tid >> 5;
  const long long ebase = (long long)blockIdx.x * TILEM;

  // ---- Stage W: stage swizzled weights into LDS (linear copy) ----
  {
    const uint4* src = reinterpret_cast<const uint4*>(wsw);
    uint4* dst = reinterpret_cast<uint4*>(ldsW);
    #pragma unroll 1
    for (int i = tid; i < WTOT / 8; i += 256)   // 8192 uint4, 32 per thread
      dst[i] = src[i];
  }

  // ---- Stage A: gather + bf16 convert + A-fragment swizzle into LDS ----
  // For an aligned 8-wide K group: lane = ((kw>>3)&1)*16 + m,
  // elements e = (kw>>4)*8 .. +7 are CONSECUTIVE -> one ds_store_b128.
  {
    const int rsub = tid >> 4;   // row within 16-row slice
    const int c16  = tid & 15;   // 8-wide K group within a 128-wide region
    #pragma unroll 1
    for (int rr = 0; rr < 8; ++rr) {
      int r = rr * 16 + rsub;            // edge row in tile; slice = rr, m = rsub
      long long e = ebase + r;
      bool valid = (e < n_edges);
      size_t bs = 0, br = 0;
      if (valid) {
        bs = (size_t)senders[e]   * DNODE;
        br = (size_t)receivers[e] * DNODE;
      }
      #pragma unroll
      for (int reg = 0; reg < 3; ++reg) {
        int k = reg * 128 + c16 * 8;     // concat column, multiple of 8
        float4 v0 = make_float4(0.f, 0.f, 0.f, 0.f);
        float4 v1 = v0;
        if (valid) {
          const float* src = (reg == 0) ? cell_attr + bs + c16 * 8
                           : (reg == 1) ? cell_attr + br + c16 * 8
                                        : edge_attr + (size_t)e * 128 + c16 * 8;
          v0 = *reinterpret_cast<const float4*>(src);
          v1 = *reinterpret_cast<const float4*>(src + 4);
        }
        int kk = k >> 5, kw = k & 31;
        int lane2 = ((kw >> 3) & 1) * 16 + rsub;
        int e0 = (kw >> 4) * 8;
        uint4 p = make_uint4(pack2(v0.x, v0.y), pack2(v0.z, v0.w),
                             pack2(v1.x, v1.y), pack2(v1.z, v1.w));
        *reinterpret_cast<uint4*>(
            &ldsA[((rr * KC1 + kk) * 32 + lane2) * 16 + e0]) = p;
      }
    }
  }
  __syncthreads();

  const int hme = lane >> 4;   // lane half
  const int nlo = lane & 15;   // N within tile

  // ---- Stage B: GEMM1  acc = X @ W1 + b1, ReLU ----
  v8f acc[NT];
  #pragma unroll
  for (int nt = 0; nt < NT; ++nt) {
    float bv = b1[nt * 16 + nlo];
    #pragma unroll
    for (int i = 0; i < 8; ++i) acc[nt][i] = bv;
  }
  // unroll capped at 2: keeps in-flight fragment set small -> no scratch spills
  #pragma unroll 2
  for (int kk = 0; kk < KC1; ++kk) {
    v16bf afrag = *reinterpret_cast<const v16bf*>(
        &ldsA[((wave * KC1 + kk) * 32 + lane) * 16]);
    #pragma unroll
    for (int nt = 0; nt < NT; ++nt) {
      v16bf bfrag = *reinterpret_cast<const v16bf*>(
          &ldsW[((kk * NT + nt) * 32 + lane) * 16]);
      acc[nt] = __builtin_amdgcn_wmma_f32_16x16x32_bf16(
          false, afrag, false, bfrag, (short)0, acc[nt], false, false);
    }
  }
  #pragma unroll
  for (int nt = 0; nt < NT; ++nt)
    #pragma unroll
    for (int i = 0; i < 8; ++i) acc[nt][i] = fmaxf(acc[nt][i], 0.f);

  // ---- Stage C: D-layout -> A-fragment transpose (wave-private LDS) ----
  // Reuse this wave's own (already consumed) A-slice as the H buffer.
  unsigned short* hbase = &ldsA[wave * KC1 * 32 * 16];
  #pragma unroll
  for (int nt = 0; nt < NT; ++nt) {
    int kidx = nt * 16 + nlo;                  // hidden index = GEMM2 K
    int kk = kidx >> 5, kw = kidx & 31;
    int h = (kw >> 3) & 1;
    int el = (kw >> 4) * 8 + (kw & 7);
    #pragma unroll
    for (int rr = 0; rr < 8; ++rr) {
      int m = rr + 8 * hme;                    // row within 16-row slice
      hbase[(kk * 32 + (h * 16 + m)) * 16 + el] =
          (unsigned short)f2bf(acc[nt][rr]);
    }
  }

  // ---- Stage D: GEMM2  acc2 = H @ W2 + b2 ----
  v8f acc2[NT];
  #pragma unroll
  for (int nt = 0; nt < NT; ++nt) {
    float bv = b2[nt * 16 + nlo];
    #pragma unroll
    for (int i = 0; i < 8; ++i) acc2[nt][i] = bv;
  }
  #pragma unroll 2
  for (int kk = 0; kk < KC2; ++kk) {
    v16bf afrag = *reinterpret_cast<const v16bf*>(&hbase[(kk * 32 + lane) * 16]);
    #pragma unroll
    for (int nt = 0; nt < NT; ++nt) {
      v16bf bfrag = *reinterpret_cast<const v16bf*>(
          &ldsW[W2OFF + ((kk * NT + nt) * 32 + lane) * 16]);
      acc2[nt] = __builtin_amdgcn_wmma_f32_16x16x32_bf16(
          false, afrag, false, bfrag, (short)0, acc2[nt], false, false);
    }
  }

  // ---- Stage E: store f32 output (lanes 0-15 cover consecutive n) ----
  #pragma unroll
  for (int nt = 0; nt < NT; ++nt) {
    #pragma unroll
    for (int rr = 0; rr < 8; ++rr) {
      long long erow = ebase + wave * 16 + (rr + 8 * hme);
      if (erow < n_edges)
        out[erow * DOUT + nt * 16 + nlo] = acc2[nt][rr];
    }
  }
}

// ---------------------------------------------------------------------------
extern "C" void kernel_launch(void* const* d_in, const int* in_sizes, int n_in,
                              void* d_out, int out_size, void* d_ws, size_t ws_size,
                              hipStream_t stream) {
  const float* cell = (const float*)d_in[0];
  const float* eatt = (const float*)d_in[1];
  const float* W1   = (const float*)d_in[2];
  const float* b1   = (const float*)d_in[3];
  const float* W2   = (const float*)d_in[4];
  const float* b2   = (const float*)d_in[5];
  const int*   eidx = (const int*)d_in[6];

  int n_edges = in_sizes[1] / 128;        // edge_attr is [E,128]
  unsigned short* ws = (unsigned short*)d_ws;

  prep_weights<<<(WTOT + 255) / 256, 256, 0, stream>>>(W1, W2, ws);

  int blocks = (n_edges + TILEM - 1) / TILEM;
  edge_mlp<<<blocks, 256, 0, stream>>>(cell, eatt, b1, b2,
                                       eidx, eidx + n_edges,
                                       ws, (float*)d_out, n_edges);
}
